// LearnableLogSinkhorn_2173253452349
// MI455X (gfx1250) — compile-verified, matrix operand order checked by reading, not run on hardware
//
#include <hip/hip_runtime.h>
#include <hip/hip_bf16.h>

// ---------------------------------------------------------------------------
// LearnableLogSinkhorn on MI455X (gfx1250).
//
// Log-domain Sinkhorn == matrix scaling on K = exp(ps):
//   w = mu / (K z) ;  z = nu / (K^T w) ;  (100 iterations, z0 = 1)
//   out = ps + log(w) + log(z) - norm
// K is iteration-constant, stored in bf16 (~72 MB, resident in 192 MB L2).
// Row pass: v_wmma_f32_16x16x32_bf16, 2 m-tiles per wave, double-buffered
// loads (loads for chunk c+1 in flight while WMMA consumes chunk c).
// Col pass: coalesced packed-bf16 (uint2 = 4 columns per lane).
// ---------------------------------------------------------------------------

typedef __attribute__((ext_vector_type(16))) __bf16 v16bf_t;
typedef __attribute__((ext_vector_type(8)))  float  v8f_t;

union ABu { uint4 q[2]; v16bf_t v; };

__device__ __forceinline__ unsigned short f2bf(float f) {
  unsigned int u = __float_as_uint(f);
  unsigned int r = u + 0x7fffu + ((u >> 16) & 1u);   // round-to-nearest-even
  return (unsigned short)(r >> 16);
}
__device__ __forceinline__ float bflo(unsigned int v) { return __uint_as_float(v << 16); }
__device__ __forceinline__ float bfhi(unsigned int v) { return __uint_as_float(v & 0xffff0000u); }

// ---------------------------------------------------------------------------
// Per-batch mask statistics: num_valid_row/col, norm = -log(r+c), e = 1/(r+c)
// ---------------------------------------------------------------------------
__global__ void batch_stats_kernel(const unsigned char* __restrict__ rm,
                                   const unsigned char* __restrict__ cm,
                                   float* nvr, float* nvc, float* normv, float* enorm,
                                   int M, int N) {
  __shared__ int sr[256], sc[256];
  int b = blockIdx.x, t = threadIdx.x;
  int cr = 0, cc = 0;
  for (int i = t; i < M; i += 256) cr += rm[(size_t)b * M + i] ? 1 : 0;
  for (int i = t; i < N; i += 256) cc += cm[(size_t)b * N + i] ? 1 : 0;
  sr[t] = cr; sc[t] = cc;
  __syncthreads();
  for (int s = 128; s > 0; s >>= 1) {
    if (t < s) { sr[t] += sr[t + s]; sc[t] += sc[t + s]; }
    __syncthreads();
  }
  if (t == 0) {
    float r = (float)sr[0], c = (float)sc[0];
    nvr[b] = r; nvc[b] = c;
    normv[b] = -logf(r + c);
    enorm[b] = 1.0f / (r + c);
  }
}

// ---------------------------------------------------------------------------
// mu (B x MP), nu (B x NP), w = 0, z = 1 on valid range (v0 = 0), 0 in padding
// ---------------------------------------------------------------------------
__global__ void init_vectors_kernel(const unsigned char* __restrict__ rm,
                                    const unsigned char* __restrict__ cm,
                                    const float* nvr, const float* nvc, const float* enorm,
                                    float* mu, float* nu, float* w32,
                                    float* z32, unsigned short* z16,
                                    int M, int N, int MP, int NP) {
  int b = blockIdx.y;
  int i = blockIdx.x * 256 + threadIdx.x;
  float e = enorm[b];
  if (i < MP) {
    float m;
    if (i < M)       m = rm[(size_t)b * M + i] ? e : 0.f;
    else if (i == M) m = nvc[b] * e;
    else             m = 0.f;
    mu[(size_t)b * MP + i]  = m;
    w32[(size_t)b * MP + i] = 0.f;
  }
  if (i < NP) {
    float nv;
    if (i < N)       nv = cm[(size_t)b * N + i] ? e : 0.f;
    else if (i == N) nv = nvr[b] * e;
    else             nv = 0.f;
    nu[(size_t)b * NP + i] = nv;
    bool live = (i <= N);
    z32[(size_t)b * NP + i] = live ? 1.f : 0.f;
    z16[(size_t)b * NP + i] = live ? (unsigned short)0x3f80 : (unsigned short)0;
  }
}

// ---------------------------------------------------------------------------
// K16[b, row, n] = valid ? exp(padded_score) : 0   (bf16, zero padding)
// ---------------------------------------------------------------------------
__global__ void build_K_kernel(const float* __restrict__ scores,
                               const float* __restrict__ alpha,
                               const unsigned char* __restrict__ rm,
                               const unsigned char* __restrict__ cm,
                               unsigned short* __restrict__ K16,
                               int M, int N, int MP, int NP) {
  int b   = blockIdx.z;
  int row = blockIdx.y;
  int n   = blockIdx.x * 64 + threadIdx.x;
  if (n >= NP) return;
  float val = 0.f; bool valid = false;
  if (row <= M && n <= N) {
    bool rv = (row < M) ? (rm[(size_t)b * M + row] != 0) : true;
    bool cv = (n   < N) ? (cm[(size_t)b * N + n]   != 0) : true;
    valid = rv && cv;
    val = (row < M && n < N) ? scores[((size_t)b * M + row) * N + n] : alpha[0];
  }
  float k = valid ? __expf(val) : 0.f;
  K16[((size_t)b * MP + row) * NP + n] = f2bf(k);
}

// ---------------------------------------------------------------------------
// Row pass: S[m] = sum_n K[m,n] * z[n] via v_wmma_f32_16x16x32_bf16,
// then w[m] = mu[m] / S[m].
// Each wave owns TWO 16-row tiles (independent accumulator chains sharing the
// z-broadcast B operand). Loads are double-buffered: chunk c+1 is issued
// before the WMMAs consuming chunk c, so waits are partial, not loadcnt==0.
// A tile = 16 rows x 32 cols of K (per lane: 2 contiguous b128 loads).
// D columns identical => lanes 0/16 hold S for rows m0..m0+7 / m0+8..15.
// ---------------------------------------------------------------------------
#define WMMA_BF16(Aop, Bop, Cop) \
  __builtin_amdgcn_wmma_f32_16x16x32_bf16(false, (Aop).v, false, (Bop).v, (short)0, (Cop), false, false)

#define LOADC(cc, Ad0, Ad1, Bd)                                              \
  {                                                                          \
    const uint4* a0p = (const uint4*)(K16 + row0 + (size_t)(cc) * 32);       \
    const uint4* a1p = (const uint4*)(K16 + row1 + (size_t)(cc) * 32);       \
    const uint4* bp  = zrow + (cc) * 4;                                      \
    (Bd).q[0]  = bp[h];   (Bd).q[1]  = bp[h + 2];                            \
    (Ad0).q[0] = a0p[h];  (Ad0).q[1] = a0p[h + 2];                           \
    (Ad1).q[0] = a1p[h];  (Ad1).q[1] = a1p[h + 2];                           \
  }

__global__ void row_pass_kernel(const unsigned short* __restrict__ K16,
                                const unsigned short* __restrict__ z16,
                                const float* __restrict__ mu,
                                float* __restrict__ w32,
                                int Mreal, int MP, int NP) {
  int lane = threadIdx.x & 31;
  int wv   = threadIdx.x >> 5;               // 4 waves per block
  int b    = blockIdx.y;
  int nt   = MP >> 4;                        // number of 16-row tiles
  int mt0  = (blockIdx.x * 4 + wv) * 2;      // first tile of this wave
  if (mt0 >= nt) return;                     // wave-uniform exit (EXEC all-1s)
  bool has2 = (mt0 + 1) < nt;
  int h = lane >> 4, ml = lane & 15;

  const size_t row0 = ((size_t)b * MP + (size_t)mt0 * 16 + ml) * NP;
  const size_t row1 = has2 ? row0 + (size_t)16 * NP : row0;  // branch-free alias
  const uint4* zrow = (const uint4*)(z16 + (size_t)b * NP);
  const int chunks = NP >> 5;                // even (NP multiple of 64)

  v8f_t acc0 = {0.f, 0.f, 0.f, 0.f, 0.f, 0.f, 0.f, 0.f};
  v8f_t acc1 = acc0;

  ABu Aa0, Aa1, Ba, Ab0, Ab1, Bb;
  LOADC(0, Aa0, Aa1, Ba);
  int c = 0;
  for (; c + 2 < chunks; c += 2) {
    LOADC(c + 1, Ab0, Ab1, Bb);              // next-chunk loads in flight
    acc0 = WMMA_BF16(Aa0, Ba, acc0);
    acc1 = WMMA_BF16(Aa1, Ba, acc1);
    LOADC(c + 2, Aa0, Aa1, Ba);
    acc0 = WMMA_BF16(Ab0, Bb, acc0);
    acc1 = WMMA_BF16(Ab1, Bb, acc1);
  }
  LOADC(c + 1, Ab0, Ab1, Bb);
  acc0 = WMMA_BF16(Aa0, Ba, acc0);
  acc1 = WMMA_BF16(Aa1, Ba, acc1);
  acc0 = WMMA_BF16(Ab0, Bb, acc0);
  acc1 = WMMA_BF16(Ab1, Bb, acc1);

  if (ml == 0) {
#pragma unroll
    for (int r = 0; r < 8; ++r) {
      int m = mt0 * 16 + h * 8 + r;          // D vgpr r: M=r (lanes 0-15), M=8+r (16-31)
      if (m < Mreal) {
        float S  = acc0[r];
        float mv = mu[(size_t)b * MP + m];
        w32[(size_t)b * MP + m] = (S > 0.f) ? (mv / S) : 0.f;
      }
    }
    if (has2) {
#pragma unroll
      for (int r = 0; r < 8; ++r) {
        int m = (mt0 + 1) * 16 + h * 8 + r;
        if (m < Mreal) {
          float S  = acc1[r];
          float mv = mu[(size_t)b * MP + m];
          w32[(size_t)b * MP + m] = (S > 0.f) ? (mv / S) : 0.f;
        }
      }
    }
  }
}

// ---------------------------------------------------------------------------
// Col pass: S'[n] = sum_m K[m,n] * w[m]  (coalesced packed-bf16 reads),
// then z[n] = nu[n] / S'[n]  (f32 + bf16 copies).
// Block: 256 threads; lane group tc (16 lanes) owns 4 adjacent columns via
// one uint2 load per row; tm (16 groups) strides the rows. 128 B contiguous
// per 16-lane half-wave per row. Partials reduced through LDS.
// ---------------------------------------------------------------------------
__global__ void col_pass_kernel(const unsigned short* __restrict__ K16,
                                const float* __restrict__ w32,
                                const float* __restrict__ nu,
                                float* __restrict__ z32,
                                unsigned short* __restrict__ z16,
                                int Nreal, int MP, int NP) {
  extern __shared__ float smem[];
  float* lw   = smem;         // MP floats
  float* part = smem + MP;    // 16 * 64 floats

  int b = blockIdx.y;
  int t = threadIdx.x;
  for (int m = t; m < MP; m += 256) lw[m] = w32[(size_t)b * MP + m];
  __syncthreads();

  int tc = t & 15;            // 16 column groups x 4 columns = 64 columns
  int tm = t >> 4;            // 16 row groups
  int n0 = blockIdx.x * 64 + tc * 4;
  const unsigned short* base = K16 + (size_t)b * MP * NP + n0;

  float a0 = 0.f, a1 = 0.f, a2 = 0.f, a3 = 0.f;
  for (int m = tm; m < MP; m += 16) {
    uint2 v = *(const uint2*)(base + (size_t)m * NP);
    float wm = lw[m];
    a0 = fmaf(bflo(v.x), wm, a0);
    a1 = fmaf(bfhi(v.x), wm, a1);
    a2 = fmaf(bflo(v.y), wm, a2);
    a3 = fmaf(bfhi(v.y), wm, a3);
  }
  part[tm * 64 + tc * 4 + 0] = a0;
  part[tm * 64 + tc * 4 + 1] = a1;
  part[tm * 64 + tc * 4 + 2] = a2;
  part[tm * 64 + tc * 4 + 3] = a3;
  __syncthreads();

  if (t < 64) {
    float s = 0.f;
#pragma unroll
    for (int k = 0; k < 16; ++k) s += part[k * 64 + t];
    int n = blockIdx.x * 64 + t;
    if (n < Nreal) {
      float nv = nu[(size_t)b * NP + n];
      float z  = (s > 0.f) ? (nv / s) : 0.f;
      z32[(size_t)b * NP + n] = z;
      z16[(size_t)b * NP + n] = f2bf(z);
    }
  }
}

// ---------------------------------------------------------------------------
// out[b,i,j] = ps + log(w[i]) + log(z[j]) - norm
// ---------------------------------------------------------------------------
__global__ void output_kernel(const float* __restrict__ scores,
                              const float* __restrict__ alpha,
                              const unsigned char* __restrict__ rm,
                              const unsigned char* __restrict__ cm,
                              const float* __restrict__ w32,
                              const float* __restrict__ z32,
                              const float* __restrict__ normv,
                              float* __restrict__ out,
                              int M, int N, int MP, int NP) {
  int b = blockIdx.z, i = blockIdx.y;
  int j = blockIdx.x * 256 + threadIdx.x;
  if (j > N) return;
  bool rv = (i < M) ? (rm[(size_t)b * M + i] != 0) : true;
  bool cv = (j < N) ? (cm[(size_t)b * N + j] != 0) : true;
  float ps;
  if (!(rv && cv))         ps = -1e12f;
  else if (i < M && j < N) ps = scores[((size_t)b * M + i) * N + j];
  else                     ps = alpha[0];
  float u = logf(w32[(size_t)b * MP + i]);
  float v = logf(z32[(size_t)b * NP + j]);
  out[((size_t)b * (M + 1) + i) * (N + 1) + j] = ps + u + v - normv[b];
}

// ---------------------------------------------------------------------------
extern "C" void kernel_launch(void* const* d_in, const int* in_sizes, int n_in,
                              void* d_out, int out_size, void* d_ws, size_t ws_size,
                              hipStream_t stream) {
  (void)n_in; (void)out_size; (void)ws_size;
  const float*         scores = (const float*)d_in[0];
  const float*         alpha  = (const float*)d_in[1];
  const unsigned char* rm     = (const unsigned char*)d_in[2];
  const unsigned char* cm     = (const unsigned char*)d_in[3];
  float*               out    = (float*)d_out;

  const int s  = in_sizes[0];      // B*M*N
  const int rs = in_sizes[2];      // B*M
  const int cs = in_sizes[3];      // B*N
  const int N  = s / rs;           // 1024
  const int M  = s / cs;           // 1024
  const int B  = rs / M;           // 32
  const int MP = ((M + 1 + 15) / 16) * 16;   // 1040 (16-row WMMA tiles)
  const int NP = ((N + 1 + 63) / 64) * 64;   // 1088 (32-col WMMA chunks, 64-col blocks)

  char* ws = (char*)d_ws;
  size_t off = 0;
  auto wsAlloc = [&](size_t bytes) -> void* {
    void* p = (void*)(ws + off);
    off += (bytes + 255) & ~(size_t)255;
    return p;
  };
  unsigned short* K16 = (unsigned short*)wsAlloc((size_t)B * MP * NP * 2);
  float*  w32  = (float*)wsAlloc((size_t)B * MP * 4);
  float*  z32  = (float*)wsAlloc((size_t)B * NP * 4);
  unsigned short* z16 = (unsigned short*)wsAlloc((size_t)B * NP * 2);
  float*  mu   = (float*)wsAlloc((size_t)B * MP * 4);
  float*  nu   = (float*)wsAlloc((size_t)B * NP * 4);
  float*  nvr  = (float*)wsAlloc((size_t)B * 4);
  float*  nvc  = (float*)wsAlloc((size_t)B * 4);
  float*  nrm  = (float*)wsAlloc((size_t)B * 4);
  float*  enrm = (float*)wsAlloc((size_t)B * 4);

  batch_stats_kernel<<<B, 256, 0, stream>>>(rm, cm, nvr, nvc, nrm, enrm, M, N);

  dim3 gv((NP + 255) / 256, B);
  init_vectors_kernel<<<gv, 256, 0, stream>>>(rm, cm, nvr, nvc, enrm,
                                              mu, nu, w32, z32, z16, M, N, MP, NP);

  dim3 gk((NP + 63) / 64, MP, B);
  build_K_kernel<<<gk, 64, 0, stream>>>(scores, alpha, rm, cm, K16, M, N, MP, NP);

  const int nt = MP / 16;                         // 16-row tiles
  const int wavesPerBatch = (nt + 1) / 2;         // 2 tiles per wave
  dim3 gr((wavesPerBatch + 3) / 4, B);            // 4 waves per block
  dim3 gc(NP / 64, B);
  size_t colShmem = (size_t)MP * 4 + 16 * 64 * 4;
  for (int it = 0; it < 100; ++it) {
    row_pass_kernel<<<gr, 128, 0, stream>>>(K16, z16, mu, w32, M + 1, MP, NP);
    col_pass_kernel<<<gc, 256, colShmem, stream>>>(K16, w32, nu, z32, z16,
                                                   N + 1, MP, NP);
  }

  dim3 go((N + 1 + 255) / 256, M + 1, B);
  output_kernel<<<go, 256, 0, stream>>>(scores, alpha, rm, cm, w32, z32, nrm,
                                        out, M, N, MP, NP);
}
